// Attention_73254962201049
// MI455X (gfx1250) — matrix operand order repaired and checked by reading, not compiled, not run on hardware
//
#include <hip/hip_runtime.h>

// ---------------- problem constants ----------------
#define DIM    1024
#define HEADS  16
#define HD     64
#define NTOK   1032
#define BATCH  8
#define NP     1056                 // NTOK padded to multiple of 32
#define MROWS  (BATCH * NTOK)       // 8256
#define MT     (MROWS / 16)         // 516 (exact)
#define NTQ    65                   // ceil(NTOK/16) query tiles
#define NKB    33                   // NP/32 key blocks
#define WINDOW 3

// ---------------- types ----------------
typedef __attribute__((ext_vector_type(16))) __bf16          v16bf;
typedef __attribute__((ext_vector_type(8)))  float           v8f;
typedef __attribute__((ext_vector_type(4)))  float           f32x4;
typedef __attribute__((ext_vector_type(8)))  unsigned short  v8us;
typedef __attribute__((ext_vector_type(16))) unsigned short  v16us;

union Frag16 { v16us u; v16bf v; };
union BfBits { __bf16 h; unsigned short s; };

// native f32 -> bf16 (RNE via hardware convert)
__device__ __forceinline__ unsigned short f2bf(float f) {
  BfBits b; b.h = (__bf16)f; return b.s;
}

// concat two contiguous-register 8-half chunks into an A/B fragment
__device__ __forceinline__ v16bf frag_concat(v8us lo, v8us hi) {
  Frag16 f;
  f.u = __builtin_shufflevector(lo, hi, 0,1,2,3,4,5,6,7,8,9,10,11,12,13,14,15);
  return f.v;
}

// 16 contiguous bf16 halfs in memory -> fragment (single 32B vector load)
__device__ __forceinline__ v16bf frag_load16(const unsigned short* p) {
  Frag16 f;
  f.u = *reinterpret_cast<const v16us*>(p);
  return f.v;
}

// 16 contiguous f32 in four f32x4 -> bf16 fragment (native converts)
__device__ __forceinline__ v16bf frag_cvt(f32x4 a, f32x4 b, f32x4 c, f32x4 d) {
  v16bf r;
  #pragma unroll
  for (int e = 0; e < 4; ++e) {
    r[e]      = (__bf16)a[e];
    r[4 + e]  = (__bf16)b[e];
    r[8 + e]  = (__bf16)c[e];
    r[12 + e] = (__bf16)d[e];
  }
  return r;
}

// ---------------- mask helpers ----------------
__device__ __forceinline__ bool attn_mask(int nq, int nk) {
  if (nq >= NTOK || nk >= NTOK) return false;     // padding
  if (nq < 8 || nk < 8) return true;              // global tokens
  int pq = nq - 8, pk = nk - 8;
  int dr = (pq >> 5) - (pk >> 5);
  int dc = (pq & 31) - (pk & 31);
  dr = dr < 0 ? -dr : dr;
  dc = dc < 0 ? -dc : dc;
  return dr <= WINDOW && dc <= WINDOW;
}

// conservative: can any (query in tile qt, key in block jb) be unmasked?
__device__ __forceinline__ bool block_needed(int qt, int jb) {
  if (jb == 0 || qt == 0) return true;            // specials present
  int q0 = qt * 16, k0 = jb * 32;
  int pq0 = q0 - 8;
  int pq1 = ((q0 + 15 < NTOK) ? q0 + 15 : NTOK - 1) - 8;
  int pk0 = k0 - 8;
  int pk1 = ((k0 + 31 < NTOK) ? k0 + 31 : NTOK - 1) - 8;
  if (pk0 > pk1 || pq0 > pq1) return false;
  int rq0 = pq0 >> 5, rq1 = pq1 >> 5;
  int rk0 = pk0 >> 5, rk1 = pk1 >> 5;
  return (rk0 <= rq1 + WINDOW) && (rk1 >= rq0 - WINDOW);
}

// ---------------- kernel 0: zero v^T padding columns ----------------
__global__ void zero_vt_pad(unsigned short* __restrict__ vtws) {
  const int PAD = NP - NTOK;                      // 24
  const int total = BATCH * HEADS * HD * PAD;     // 196608
  int i = blockIdx.x * 256 + threadIdx.x;
  if (i >= total) return;
  int row = i / PAD, c = i - row * PAD;
  vtws[(size_t)row * NP + NTOK + c] = 0;
}

// ---------------- kernel 1: QKV GEMM + RMSNorm + RoPE ----------------
// grid = (516, 48); blockIdx.y = mat*16 + head; 4 waves -> 16x64 tile = one head
__global__ __launch_bounds__(128)
void qkv_rope_kernel(const float* __restrict__ hidden,
                     const float* __restrict__ qkv_w,
                     const float* __restrict__ fcos,
                     const float* __restrict__ fsin,
                     const float* __restrict__ nqw,
                     const float* __restrict__ nkw,
                     unsigned short* __restrict__ qws,
                     unsigned short* __restrict__ kws,
                     unsigned short* __restrict__ vtws)
{
  __shared__ float tile[16][68];
  __shared__ float rstd[16];

  const int tid   = threadIdx.x;
  const int lane  = tid & 31;
  const int wave  = tid >> 5;
  const int l16   = lane & 15;
  const int khalf = lane >> 4;
  const int mtile = blockIdx.x;
  const int mat   = blockIdx.y >> 4;   // 0=q 1=k 2=v
  const int h     = blockIdx.y & 15;

  const int    jcol = wave * 16 + l16;                       // 0..63 within head
  const float* arow = hidden + (size_t)(mtile * 16 + l16) * DIM;
  const float* brow = qkv_w  + (size_t)(mat * DIM + h * HD + jcol) * DIM;

  v8f acc = {};
  for (int kb = 0; kb < DIM; kb += 32) {
    __builtin_prefetch(arow + kb + 128, 0, 0);
    __builtin_prefetch(brow + kb + 128, 0, 0);
    const f32x4* pa0 = reinterpret_cast<const f32x4*>(arow + kb + khalf * 16);
    const f32x4* pb  = reinterpret_cast<const f32x4*>(brow + kb + khalf * 16);
    v16bf av = frag_cvt(pa0[0], pa0[1], pa0[2], pa0[3]);
    v16bf bv = frag_cvt(pb[0], pb[1], pb[2], pb[3]);
    acc = __builtin_amdgcn_wmma_f32_16x16x32_bf16(false, av, false, bv,
                                                  (short)0, acc, false, false);
  }

  // C layout: lane col = jcol, rows = r + 8*khalf
  #pragma unroll
  for (int r = 0; r < 8; ++r) tile[r + 8 * khalf][jcol] = acc[r];
  __syncthreads();

  const int gm0 = mtile * 16;

  if (mat == 2) {                                   // V: transposed bf16 store
    const int b0  = gm0 / NTOK;
    const int b15 = (gm0 + 15) / NTOK;
    if (b0 == b15) {
      // fast path: all 16 tokens in one batch -> 8-token vector stores
      int d = tid >> 1, t0 = (tid & 1) * 8;         // 64 dims x 2 chunks
      int n0 = gm0 - b0 * NTOK;
      v8us ov;
      #pragma unroll
      for (int e = 0; e < 8; ++e) ov[e] = f2bf(tile[t0 + e][d]);
      *reinterpret_cast<v8us*>(
          vtws + ((size_t)(b0 * HEADS + h) * HD + d) * NP + n0 + t0) = ov;
    } else {
      for (int i = tid; i < 16 * 64; i += 128) {
        int m = i >> 6, d = i & 63;
        int gm = gm0 + m;
        int b  = gm / NTOK, n = gm - b * NTOK;
        vtws[((size_t)(b * HEADS + h) * HD + d) * NP + n] = f2bf(tile[m][d]);
      }
    }
    return;
  }

  if (tid < 16) {                                   // per-row RMS over 64 dims
    float s = 0.f;
    #pragma unroll 8
    for (int d = 0; d < HD; ++d) { float x = tile[tid][d]; s += x * x; }
    rstd[tid] = rsqrtf(s * (1.0f / HD) + 1e-6f);
  }
  __syncthreads();

  const float*    nw  = (mat == 0) ? nqw : nkw;
  unsigned short* dst = (mat == 0) ? qws : kws;
  {
    const int m  = tid >> 3;                        // 0..15 token row
    const int d0 = (tid & 7) * 8;                   // 8-dim chunk
    int gm = gm0 + m;
    int b  = gm / NTOK, n = gm - b * NTOK;
    float rs  = rstd[m];
    float sgn = (d0 < 32) ? -1.0f : 1.0f;           // rotate-half sign
    const float* cosp = fcos + (size_t)n * HD + d0;
    const float* sinp = fsin + (size_t)n * HD + d0;
    v8us outv;
    #pragma unroll
    for (int e = 0; e < 8; ++e) {
      int d  = d0 + e;
      int d2 = (d + 32) & 63;
      float xn  = tile[m][d] * rs * nw[d];
      float rot = sgn * tile[m][d2] * rs * nw[d2];
      outv[e] = f2bf(xn * cosp[e] + rot * sinp[e]);
    }
    *reinterpret_cast<v8us*>(
        dst + ((size_t)(b * HEADS + h) * NP + n) * HD + d0) = outv;
  }
}

// ---------------- kernel 2: masked flash attention ----------------
// grid = (65, HEADS, BATCH), one wave per block; 16 queries, online softmax
__global__ __launch_bounds__(32)
void attn_kernel(const unsigned short* __restrict__ qws,
                 const unsigned short* __restrict__ kws,
                 const unsigned short* __restrict__ vtws,
                 unsigned short* __restrict__ attnws)
{
  __shared__ unsigned short pbuf[16][32];

  const int lane  = threadIdx.x;
  const int l16   = lane & 15;
  const int khalf = lane >> 4;
  const int qt = blockIdx.x, h = blockIdx.y, b = blockIdx.z;
  const int bh = b * HEADS + h;

  // Q fragments (A layout): kd 0..31 and 32..63
  v16bf a0, a1;
  {
    const unsigned short* qrow = qws + ((size_t)bh * NP + qt * 16 + l16) * HD;
    v8us x0 = *reinterpret_cast<const v8us*>(qrow + khalf * 16);
    v8us x1 = *reinterpret_cast<const v8us*>(qrow + 8 + khalf * 16);
    v8us x2 = *reinterpret_cast<const v8us*>(qrow + 32 + khalf * 16);
    v8us x3 = *reinterpret_cast<const v8us*>(qrow + 40 + khalf * 16);
    a0 = frag_concat(x0, x1);
    a1 = frag_concat(x2, x3);
  }

  v8f o0 = {}, o1 = {}, o2 = {}, o3 = {};
  float mr[8], lr[8];
  int   nq[8];
  #pragma unroll
  for (int r = 0; r < 8; ++r) {
    mr[r] = -__builtin_inff(); lr[r] = 0.0f;
    nq[r] = qt * 16 + r + 8 * khalf;
  }

  const unsigned short* kbase = kws  + (size_t)bh * NP * HD;
  const unsigned short* vbase = vtws + (size_t)bh * HD * NP;
  const float scale = 0.125f;                      // 1/sqrt(64)

  for (int jb = 0; jb < NKB; ++jb) {
    if (!block_needed(qt, jb)) continue;
    const int k0 = jb * 32;

    // ---- S = Q K^T for 32 keys (two 16-key C tiles); B-frags contiguous ----
    v8f s0 = {}, s1 = {};
    {
      const unsigned short* kr0 = kbase + (size_t)(k0 + l16) * HD + khalf * 16;
      const unsigned short* kr1 = kr0 + (size_t)16 * HD;
      s0 = __builtin_amdgcn_wmma_f32_16x16x32_bf16(false, a0, false,
              frag_load16(kr0),      (short)0, s0, false, false);
      s0 = __builtin_amdgcn_wmma_f32_16x16x32_bf16(false, a1, false,
              frag_load16(kr0 + 32), (short)0, s0, false, false);
      s1 = __builtin_amdgcn_wmma_f32_16x16x32_bf16(false, a0, false,
              frag_load16(kr1),      (short)0, s1, false, false);
      s1 = __builtin_amdgcn_wmma_f32_16x16x32_bf16(false, a1, false,
              frag_load16(kr1 + 32), (short)0, s1, false, false);
    }

    // ---- mask + online softmax (rows spread over 8 accum slots) ----
    const int nk0 = k0 + l16, nk1 = nk0 + 16;
    float p0[8], p1[8];
    #pragma unroll
    for (int r = 0; r < 8; ++r) {
      float x0 = attn_mask(nq[r], nk0) ? s0[r] * scale : -__builtin_inff();
      float x1 = attn_mask(nq[r], nk1) ? s1[r] * scale : -__builtin_inff();
      float t = fmaxf(x0, x1);
      #pragma unroll
      for (int off = 1; off < 16; off <<= 1) t = fmaxf(t, __shfl_xor(t, off, 32));
      float mn = fmaxf(mr[r], t);
      float sc = __expf(mr[r] - mn);
      float e0 = __expf(x0 - mn);
      float e1 = __expf(x1 - mn);
      float rs = e0 + e1;
      #pragma unroll
      for (int off = 1; off < 16; off <<= 1) rs += __shfl_xor(rs, off, 32);
      lr[r] = lr[r] * sc + rs;
      mr[r] = mn;
      o0[r] *= sc; o1[r] *= sc; o2[r] *= sc; o3[r] *= sc;
      p0[r] = e0;  p1[r] = e1;
    }

    // ---- re-layout P (C layout -> A layout) through LDS ----
    __syncthreads();
    #pragma unroll
    for (int r = 0; r < 8; ++r) {
      int m = r + 8 * khalf;
      pbuf[m][l16]      = f2bf(p0[r]);
      pbuf[m][16 + l16] = f2bf(p1[r]);
    }
    __syncthreads();
    v16bf pa;
    {
      v8us lo = *reinterpret_cast<const v8us*>(&pbuf[l16][khalf * 16]);
      v8us hi = *reinterpret_cast<const v8us*>(&pbuf[l16][8 + khalf * 16]);
      pa = frag_concat(lo, hi);
    }

    // ---- O += P V  (v transposed: 16 contiguous halfs per B-frag lane) ----
    {
      const unsigned short* vp = vbase + (size_t)l16 * NP + k0 + khalf * 16;
      o0 = __builtin_amdgcn_wmma_f32_16x16x32_bf16(false, pa, false,
              frag_load16(vp),                (short)0, o0, false, false);
      o1 = __builtin_amdgcn_wmma_f32_16x16x32_bf16(false, pa, false,
              frag_load16(vp + 16 * NP),      (short)0, o1, false, false);
      o2 = __builtin_amdgcn_wmma_f32_16x16x32_bf16(false, pa, false,
              frag_load16(vp + 32 * NP),      (short)0, o2, false, false);
      o3 = __builtin_amdgcn_wmma_f32_16x16x32_bf16(false, pa, false,
              frag_load16(vp + 48 * NP),      (short)0, o3, false, false);
    }
  }

  // ---- normalize + store bf16 to attention workspace [B*NTOK, DIM] ----
  #pragma unroll
  for (int r = 0; r < 8; ++r) {
    int n = nq[r];
    if (n >= NTOK) continue;
    float inv = 1.0f / lr[r];
    size_t row = (size_t)(b * NTOK + n) * DIM + h * HD;
    attnws[row +  0 + l16] = f2bf(o0[r] * inv);
    attnws[row + 16 + l16] = f2bf(o1[r] * inv);
    attnws[row + 32 + l16] = f2bf(o2[r] * inv);
    attnws[row + 48 + l16] = f2bf(o3[r] * inv);
  }
}

// ---------------- kernel 3: output projection ----------------
// grid = (516, 16), 4 waves each own a 16x16 tile; out = attn @ out_w^T (f32)
__global__ __launch_bounds__(128)
void out_proj_kernel(const unsigned short* __restrict__ attnws,
                     const float* __restrict__ out_w,
                     float* __restrict__ out)
{
  const int tid   = threadIdx.x;
  const int lane  = tid & 31;
  const int wave  = tid >> 5;
  const int l16   = lane & 15;
  const int khalf = lane >> 4;
  const int mtile = blockIdx.x;
  const int jcol  = blockIdx.y * 64 + wave * 16 + l16;

  const unsigned short* arow = attnws + (size_t)(mtile * 16 + l16) * DIM;
  const float*          brow = out_w + (size_t)jcol * DIM;

  v8f acc = {};
  for (int kb = 0; kb < DIM; kb += 32) {
    __builtin_prefetch(brow + kb + 128, 0, 0);
    v8us lo = *reinterpret_cast<const v8us*>(arow + kb + khalf * 16);
    v8us hi = *reinterpret_cast<const v8us*>(arow + kb + 8 + khalf * 16);
    v16bf av = frag_concat(lo, hi);
    const f32x4* pb = reinterpret_cast<const f32x4*>(brow + kb + khalf * 16);
    v16bf bv = frag_cvt(pb[0], pb[1], pb[2], pb[3]);
    acc = __builtin_amdgcn_wmma_f32_16x16x32_bf16(false, av, false, bv,
                                                  (short)0, acc, false, false);
  }
  #pragma unroll
  for (int r = 0; r < 8; ++r)
    out[(size_t)(mtile * 16 + r + 8 * khalf) * DIM + jcol] = acc[r];
}

// ---------------- launcher ----------------
extern "C" void kernel_launch(void* const* d_in, const int* in_sizes, int n_in,
                              void* d_out, int out_size, void* d_ws, size_t ws_size,
                              hipStream_t stream) {
  (void)in_sizes; (void)n_in; (void)out_size; (void)ws_size;
  const float* hidden = (const float*)d_in[0];
  const float* fcos   = (const float*)d_in[1];
  const float* fsin   = (const float*)d_in[2];
  const float* qkv_w  = (const float*)d_in[3];
  const float* out_w  = (const float*)d_in[4];
  const float* nqw    = (const float*)d_in[5];
  const float* nkw    = (const float*)d_in[6];
  float* out = (float*)d_out;

  const size_t QKV_ELEMS = (size_t)BATCH * HEADS * NP * HD;   // 8.65M bf16 each
  unsigned short* qws  = (unsigned short*)d_ws;
  unsigned short* kws  = qws + QKV_ELEMS;
  unsigned short* vtws = kws + QKV_ELEMS;
  unsigned short* aws  = vtws + QKV_ELEMS;                    // [8256,1024] bf16

  {
    const int total = BATCH * HEADS * HD * (NP - NTOK);
    zero_vt_pad<<<dim3((total + 255) / 256), dim3(256), 0, stream>>>(vtws);
  }
  qkv_rope_kernel<<<dim3(MT, 3 * HEADS), dim3(128), 0, stream>>>(
      hidden, qkv_w, fcos, fsin, nqw, nkw, qws, kws, vtws);
  attn_kernel<<<dim3(NTQ, HEADS, BATCH), dim3(32), 0, stream>>>(
      qws, kws, vtws, aws);
  out_proj_kernel<<<dim3(MT, DIM / 64), dim3(128), 0, stream>>>(
      aws, out_w, out);
}
